// GraphAttentionNetworkTransductive3_6648609374463
// MI455X (gfx1250) — compile-verified, matrix operand order checked by reading, not run on hardware
//
#include <hip/hip_runtime.h>
#include <hip/hip_bf16.h>
#include <stdint.h>

// ---------------------------------------------------------------------------
// Problem constants (from reference)
// ---------------------------------------------------------------------------
#define GN      50000     // nodes
#define GE      400000    // edges
#define F_IN    1024
#define GD      512       // D == HEADS*U
#define HEADS   4
#define GU      128
#define GFF     2048
#define G_OUT   40
#define G_L     4
#define N_IDX   1000
#define NEG_SLOPE 0.2f
#define LN_EPS    0.001f

typedef __attribute__((ext_vector_type(16))) __bf16 v16bf;
typedef __attribute__((ext_vector_type(8)))  __bf16 bf16x8;
typedef __attribute__((ext_vector_type(4)))  __bf16 bf16x4;
typedef __attribute__((ext_vector_type(8)))  float  v8f;

union V16U { v16bf v; bf16x8 h[2]; };

static __device__ __forceinline__ __bf16 f2bf(float f) {
    union { float f; uint32_t u; } c; c.f = f;
    uint32_t u = c.u;
    uint32_t r = (u + 0x7FFFu + ((u >> 16) & 1u)) >> 16;   // round-to-nearest-even
    union { uint16_t s; __bf16 b; } o; o.s = (uint16_t)r;
    return o.b;
}

static __device__ __forceinline__ uint32_t encf(float f) {
    union { float f; uint32_t u; } c; c.f = f;
    return (c.u & 0x80000000u) ? ~c.u : (c.u | 0x80000000u);
}
static __device__ __forceinline__ float decf(uint32_t k) {
    union { uint32_t u; float f; } c;
    c.u = (k & 0x80000000u) ? (k ^ 0x80000000u) : ~k;
    return c.f;
}

static __device__ __forceinline__ float gelu_tanh(float x) {
    float x3 = x * x * x;
    return 0.5f * x * (1.0f + tanhf(0.7978845608028654f * (x + 0.044715f * x3)));
}

// LDS byte offset of a generic pointer into __shared__ (aperture rule:
// flat shared address low 32 bits == offset within workgroup LDS alloc).
static __device__ __forceinline__ unsigned lds_off32(const void* p) {
    return (unsigned)(unsigned long long)p;
}

// async copy of 16 contiguous bytes: global (base + voff) -> LDS at ldsoff
static __device__ __forceinline__ void async_b128(unsigned ldsoff, unsigned voff,
                                                  unsigned long long base) {
    asm volatile("global_load_async_to_lds_b128 %0, %1, %2"
                 :: "v"(ldsoff), "v"(voff), "s"(base) : "memory");
}

static __device__ __forceinline__ void wait_async_le0() {
#if __has_builtin(__builtin_amdgcn_s_wait_asynccnt)
    __builtin_amdgcn_s_wait_asynccnt(0);
#else
    asm volatile("s_wait_asynccnt 0x0" ::: "memory");
#endif
}
static __device__ __forceinline__ void wait_async_le6() {
#if __has_builtin(__builtin_amdgcn_s_wait_asynccnt)
    __builtin_amdgcn_s_wait_asynccnt(6);
#else
    asm volatile("s_wait_asynccnt 0x6" ::: "memory");
#endif
}

// ---------------------------------------------------------------------------
// WMMA bf16 GEMM:  C[M,Ncols](fp32) = A[M,K](bf16,row-major) * Bt[Ncols,K](bf16)
// + bias. Block tile 128x64, K step 64, 8 waves (4x2), each wave 32x32.
// Double-buffered LDS staging via GLOBAL_LOAD_ASYNC_TO_LDS_B128 (ASYNCcnt):
// tile i+1 loads overlap tile i WMMA compute (in-order async completion lets
// s_wait_asynccnt 6 guarantee tile i is resident).
// mode 0: C[row*Ncols+col] = acc + bias[col]
// mode 1: atomicAdd(&C[row*512 + (col&511)], gelu(acc + bias[col]))  (FFN fuse)
// ---------------------------------------------------------------------------
__global__ __launch_bounds__(256) void gemm_bf16_kernel(
    const __bf16* __restrict__ A, const __bf16* __restrict__ Bt,
    const float* __restrict__ bias, float* __restrict__ C,
    int M, int K, int Ncols, int mode)
{
    __shared__ __align__(16) __bf16 As[2][128 * 64];   // 2 x 16 KB, rows of 64
    __shared__ __align__(16) __bf16 Bs[2][64 * 64];    // 2 x  8 KB, cols of 64

    const int t    = threadIdx.x;
    const int lane = t & 31;
    const int wid  = t >> 5;
    const int wm   = wid >> 1;          // 0..3  (rows of 32)
    const int wn   = wid & 1;           // 0..1  (cols of 32)
    const int hl   = lane >> 4;         // lane half
    const int lr   = lane & 15;

    const int gm0 = blockIdx.y * 128;
    const int gn0 = blockIdx.x * 64;

    const unsigned long long abase = (unsigned long long)(const void*)A;
    const unsigned long long bbase = (unsigned long long)(const void*)Bt;

    v8f acc[2][2] = {};

    // stage one 128x64 A tile + 64x64 B tile (6 async b128 issues per thread;
    // OOB A rows are clamped - their results are discarded in the epilogue)
    auto stage = [&](int k0, __bf16* Ad, __bf16* Bd) {
        #pragma unroll
        for (int i = 0; i < 4; ++i) {
            int c    = i * 256 + t;
            int row  = c >> 3;          // 8 chunks per 128B row
            int part = c & 7;
            int grow = gm0 + row;
            if (grow >= M) grow = M - 1;           // clamp, no exec divergence
            unsigned voff = (unsigned)(((unsigned)grow * (unsigned)K
                                        + (unsigned)(k0 + part * 8)) * 2u);
            async_b128(lds_off32(Ad + row * 64 + part * 8), voff, abase);
        }
        #pragma unroll
        for (int i = 0; i < 2; ++i) {
            int c    = i * 256 + t;
            int col  = c >> 3;
            int part = c & 7;
            unsigned voff = (unsigned)(((unsigned)(gn0 + col) * (unsigned)K
                                        + (unsigned)(k0 + part * 8)) * 2u);
            async_b128(lds_off32(Bd + col * 64 + part * 8), voff, bbase);
        }
    };

    const int nk = K >> 6;
    stage(0, As[0], Bs[0]);

    for (int it = 0; it < nk; ++it) {
        const int cur = it & 1;
        if (it + 1 < nk) {
            stage((it + 1) << 6, As[cur ^ 1], Bs[cur ^ 1]);
            wait_async_le6();           // only next tile's 6 still in flight
        } else {
            wait_async_le0();
        }
        __syncthreads();                // all waves' tile `it` data resident

        #pragma unroll
        for (int ks = 0; ks < 2; ++ks) {
            const int kb = ks * 32;
            V16U afrag[2], bfrag[2];
            #pragma unroll
            for (int mi = 0; mi < 2; ++mi) {
                const __bf16* ap = &As[cur][(wm * 32 + mi * 16 + lr) * 64 + kb + hl * 8];
                afrag[mi].h[0] = *(const bf16x8*)(ap);
                afrag[mi].h[1] = *(const bf16x8*)(ap + 16);
            }
            #pragma unroll
            for (int ni = 0; ni < 2; ++ni) {
                const __bf16* bp = &Bs[cur][(wn * 32 + ni * 16 + lr) * 64 + kb + hl * 8];
                bfrag[ni].h[0] = *(const bf16x8*)(bp);
                bfrag[ni].h[1] = *(const bf16x8*)(bp + 16);
            }
            #pragma unroll
            for (int mi = 0; mi < 2; ++mi)
                #pragma unroll
                for (int ni = 0; ni < 2; ++ni)
                    acc[mi][ni] = __builtin_amdgcn_wmma_f32_16x16x32_bf16(
                        false, afrag[mi].v, false, bfrag[ni].v,
                        (short)0, acc[mi][ni], false, false);
        }
        __syncthreads();                // done reading buf before it's re-staged
    }

    // ---- epilogue ----
    #pragma unroll
    for (int mi = 0; mi < 2; ++mi) {
        #pragma unroll
        for (int ni = 0; ni < 2; ++ni) {
            int colg = gn0 + wn * 32 + ni * 16 + lr;
            float bv = bias[colg];
            #pragma unroll
            for (int j = 0; j < 8; ++j) {
                int rowg = gm0 + wm * 32 + mi * 16 + hl * 8 + j;
                if (rowg < M) {
                    float v = acc[mi][ni][j] + bv;
                    if (mode == 0) {
                        C[(size_t)rowg * Ncols + colg] = v;
                    } else {
                        atomicAdd(&C[(size_t)rowg * GD + (colg & (GD - 1))],
                                  gelu_tanh(v));
                    }
                }
            }
        }
    }
}

// ---------------------------------------------------------------------------
// fp32 -> bf16 bulk convert (8 elems / thread)
// ---------------------------------------------------------------------------
__global__ __launch_bounds__(256) void convert_bf16_kernel(
    const float* __restrict__ in, __bf16* __restrict__ out, size_t n8)
{
    size_t i = (size_t)blockIdx.x * 256 + threadIdx.x;
    if (i >= n8) return;
    const float4* p = (const float4*)in + i * 2;
    float4 a = p[0], b = p[1];
    bf16x8 r = { f2bf(a.x), f2bf(a.y), f2bf(a.z), f2bf(a.w),
                 f2bf(b.x), f2bf(b.y), f2bf(b.z), f2bf(b.w) };
    *((bf16x8*)out + i) = r;
}

// fp32 [K,N] row-major -> bf16 [N,K] (pre-transposed weights)
__global__ __launch_bounds__(256) void transpose_bf16_kernel(
    const float* __restrict__ in, __bf16* __restrict__ out,
    int K, int Ncols, size_t total)
{
    size_t i = (size_t)blockIdx.x * 256 + threadIdx.x;
    if (i >= total) return;
    int n = (int)(i / (size_t)K);
    int k = (int)(i % (size_t)K);
    out[i] = f2bf(in[(size_t)k * Ncols + n]);
}

// ---------------------------------------------------------------------------
// Layernorm: one wave per row of 512. Writes fp32 out1 (+out2 if non-null)
// and bf16 outb (GEMM operand).
// ---------------------------------------------------------------------------
__global__ __launch_bounds__(256) void layernorm_kernel(
    const float* __restrict__ in, float* __restrict__ out1, float* __restrict__ out2,
    __bf16* __restrict__ outb,
    const float* __restrict__ g, const float* __restrict__ b, int M)
{
    const int wid  = threadIdx.x >> 5;
    const int lane = threadIdx.x & 31;
    const int row  = blockIdx.x * 8 + wid;
    if (row >= M) return;

    const float4* rp = (const float4*)(in + (size_t)row * GD);
    float4 v[4];
    float s = 0.f, sq = 0.f;
    #pragma unroll
    for (int i = 0; i < 4; ++i) {
        v[i] = rp[lane + 32 * i];
        s  += v[i].x + v[i].y + v[i].z + v[i].w;
        sq += v[i].x * v[i].x + v[i].y * v[i].y + v[i].z * v[i].z + v[i].w * v[i].w;
    }
    #pragma unroll
    for (int m = 1; m < 32; m <<= 1) {
        s  += __shfl_xor(s,  m, 32);
        sq += __shfl_xor(sq, m, 32);
    }
    float mean = s * (1.0f / GD);
    float var  = sq * (1.0f / GD) - mean * mean;
    float rs   = rsqrtf(var + LN_EPS);

    float4*  o1 = (float4*)(out1 + (size_t)row * GD);
    float4*  o2 = out2 ? (float4*)(out2 + (size_t)row * GD) : nullptr;
    bf16x4*  ob = (bf16x4*)(outb + (size_t)row * GD);
    #pragma unroll
    for (int i = 0; i < 4; ++i) {
        int col = (lane + 32 * i) * 4;
        float4 r;
        r.x = (v[i].x - mean) * rs * g[col + 0] + b[col + 0];
        r.y = (v[i].y - mean) * rs * g[col + 1] + b[col + 1];
        r.z = (v[i].z - mean) * rs * g[col + 2] + b[col + 2];
        r.w = (v[i].w - mean) * rs * g[col + 3] + b[col + 3];
        o1[lane + 32 * i] = r;
        if (o2) o2[lane + 32 * i] = r;
        bf16x4 rb = { f2bf(r.x), f2bf(r.y), f2bf(r.z), f2bf(r.w) };
        ob[lane + 32 * i] = rb;
    }
}

// ---------------------------------------------------------------------------
// GATv2 edge scores: one wave per edge. lane covers d = lane*16 .. +15
// (single head per lane: head = lane/8). Butterfly reduce over 8 lanes.
// ---------------------------------------------------------------------------
__global__ __launch_bounds__(256) void edge_scores_kernel(
    const float* __restrict__ h, const int* __restrict__ edges,
    const float* __restrict__ a, float* __restrict__ scores,
    unsigned int* __restrict__ menc, int E)
{
    const int wid  = threadIdx.x >> 5;
    const int lane = threadIdx.x & 31;
    const int e    = blockIdx.x * 8 + wid;
    if (e >= E) return;
    const int tgt = edges[e * 2 + 0];
    const int src = edges[e * 2 + 1];

    const float4* ht = (const float4*)(h + (size_t)tgt * GD) + lane * 4;
    const float4* hs = (const float4*)(h + (size_t)src * GD) + lane * 4;

    const int hh = lane >> 3;
    const int u0 = (lane & 7) * 16;
    const float* aw = a + hh * GU + u0;

    float acc = 0.f;
    #pragma unroll
    for (int i = 0; i < 4; ++i) {
        float4 t4 = ht[i];
        float4 s4 = hs[i];
        float f0 = t4.x + s4.x, f1 = t4.y + s4.y, f2 = t4.z + s4.z, f3 = t4.w + s4.w;
        f0 = f0 > 0.f ? f0 : NEG_SLOPE * f0;
        f1 = f1 > 0.f ? f1 : NEG_SLOPE * f1;
        f2 = f2 > 0.f ? f2 : NEG_SLOPE * f2;
        f3 = f3 > 0.f ? f3 : NEG_SLOPE * f3;
        acc += f0 * aw[i * 4 + 0] + f1 * aw[i * 4 + 1] + f2 * aw[i * 4 + 2] + f3 * aw[i * 4 + 3];
    }
    #pragma unroll
    for (int m = 1; m < 8; m <<= 1) acc += __shfl_xor(acc, m, 32);

    if ((lane & 7) == 0) {
        scores[(size_t)e * HEADS + hh] = acc;
        atomicMax(&menc[(size_t)tgt * HEADS + hh], encf(acc));
    }
}

// ---------------------------------------------------------------------------
// ex = exp(score - max); denom += ex  (in-place on scores)
// ---------------------------------------------------------------------------
__global__ __launch_bounds__(256) void edge_exp_kernel(
    float* __restrict__ scores, const int* __restrict__ edges,
    const unsigned int* __restrict__ menc, float* __restrict__ denom, int E)
{
    int i = blockIdx.x * blockDim.x + threadIdx.x;
    if (i >= E * HEADS) return;
    int e  = i >> 2;
    int hh = i & 3;
    int tgt = edges[e * 2 + 0];
    float m = decf(menc[(size_t)tgt * HEADS + hh]);
    float ex = expf(scores[i] - m);
    scores[i] = ex;
    atomicAdd(&denom[(size_t)tgt * HEADS + hh], ex);
}

// ---------------------------------------------------------------------------
// attn[tgt] += (ex/denom[tgt]) * h[src] : one wave per edge
// ---------------------------------------------------------------------------
__global__ __launch_bounds__(256) void edge_scatter_kernel(
    const float* __restrict__ h, const int* __restrict__ edges,
    const float* __restrict__ ex, const float* __restrict__ denom,
    float* __restrict__ attn, int E)
{
    const int wid  = threadIdx.x >> 5;
    const int lane = threadIdx.x & 31;
    const int e    = blockIdx.x * 8 + wid;
    if (e >= E) return;
    const int tgt = edges[e * 2 + 0];
    const int src = edges[e * 2 + 1];
    const int hh  = lane >> 3;

    float alpha = ex[(size_t)e * HEADS + hh] / denom[(size_t)tgt * HEADS + hh];

    const float4* hs = (const float4*)(h + (size_t)src * GD) + lane * 4;
    float* dst = attn + (size_t)tgt * GD + lane * 16;
    #pragma unroll
    for (int i = 0; i < 4; ++i) {
        float4 s4 = hs[i];
        atomicAdd(dst + i * 4 + 0, alpha * s4.x);
        atomicAdd(dst + i * 4 + 1, alpha * s4.y);
        atomicAdd(dst + i * 4 + 2, alpha * s4.z);
        atomicAdd(dst + i * 4 + 3, alpha * s4.w);
    }
}

// ---------------------------------------------------------------------------
// helpers
// ---------------------------------------------------------------------------
__global__ __launch_bounds__(256) void fill_zero_kernel(float* __restrict__ p, size_t n) {
    size_t i = (size_t)blockIdx.x * blockDim.x + threadIdx.x;
    if (i < n) p[i] = 0.0f;
}

__global__ __launch_bounds__(256) void add_kernel(
    const float* __restrict__ a, const float* __restrict__ b,
    float* __restrict__ c, size_t n)
{
    size_t i = (size_t)blockIdx.x * blockDim.x + threadIdx.x;
    if (i < n) c[i] = a[i] + b[i];
}

// tail: out[j,o] = x[idx[j]] . tail_W[:,o] + tail_b[o]
__global__ __launch_bounds__(64) void tail_kernel(
    const float* __restrict__ x, const int* __restrict__ indices,
    const float* __restrict__ W, const float* __restrict__ b,
    float* __restrict__ out)
{
    int j = blockIdx.x;
    int o = threadIdx.x;
    if (o >= G_OUT) return;
    int idx = indices[j];
    const float* xr = x + (size_t)idx * GD;
    float acc = b[o];
    for (int d = 0; d < GD; ++d) acc += xr[d] * W[d * G_OUT + o];
    out[(size_t)j * G_OUT + o] = acc;
}

// ---------------------------------------------------------------------------
// launch
// ---------------------------------------------------------------------------
extern "C" void kernel_launch(void* const* d_in, const int* in_sizes, int n_in,
                              void* d_out, int out_size, void* d_ws, size_t ws_size,
                              hipStream_t stream) {
    const float* node_states = (const float*)d_in[0];
    const int*   edges       = (const int*)  d_in[1];
    const int*   indices     = (const int*)  d_in[2];
    const float* head_W      = (const float*)d_in[3];
    const float* head_b      = (const float*)d_in[4];
    const float* normg_g     = (const float*)d_in[5];
    const float* normg_b     = (const float*)d_in[6];
    const float* gat_W       = (const float*)d_in[7];
    const float* gat_a       = (const float*)d_in[8];
    const float* gat_b       = (const float*)d_in[9];
    const float* normd_g     = (const float*)d_in[10];
    const float* normd_b     = (const float*)d_in[11];
    const float* dense_W     = (const float*)d_in[12];
    const float* dense_b     = (const float*)d_in[13];
    const float* tail_W      = (const float*)d_in[14];
    const float* tail_b      = (const float*)d_in[15];
    float* out = (float*)d_out;

    // workspace partition
    float* ws = (float*)d_ws;
    const size_t nD = (size_t)GN * GD;
    float* x      = ws;                                    // [N,512]
    float* xg     = x + nD;                                // [N,512]
    float* hbuf   = xg + nD;                               // [N,512]
    float* attn   = hbuf + nD;                             // [N,512]
    float* denom  = attn + nD;                             // [N,4]
    float* mencf  = denom + (size_t)GN * HEADS;            // [N,4] uint-encoded
    float* scores = mencf + (size_t)GN * HEADS;            // [E,4]
    unsigned int* menc = (unsigned int*)mencf;
    __bf16* node_bf = (__bf16*)(scores + (size_t)GE * HEADS); // [N,1024] bf16
    __bf16* xg_bf   = node_bf + (size_t)GN * F_IN;            // [N,512]  bf16
    __bf16* headWt  = xg_bf + nD;                             // [512,1024] bf16
    __bf16* gatWt   = headWt + (size_t)GD * F_IN;             // [L][512,512]
    __bf16* denseWt = gatWt + (size_t)G_L * GD * GD;          // [L][2048,512]

    const dim3 blk(256);
    const int  gmM = (GN + 127) / 128;

    // ---- one-time conversions (bf16 activations, pre-transposed bf16 weights)
    {
        size_t n8 = (size_t)GN * F_IN / 8;
        convert_bf16_kernel<<<(int)((n8 + 255) / 256), blk, 0, stream>>>(
            node_states, node_bf, n8);
        size_t th = (size_t)GD * F_IN;   // head_W: K=1024, N=512 -> Bt[512,1024]
        transpose_bf16_kernel<<<(int)((th + 255) / 256), blk, 0, stream>>>(
            head_W, headWt, F_IN, GD, th);
        for (int l = 0; l < G_L; ++l) {
            size_t tg = (size_t)GD * GD; // gat_W[l]: K=512, N=512
            transpose_bf16_kernel<<<(int)((tg + 255) / 256), blk, 0, stream>>>(
                gat_W + (size_t)l * GD * GD, gatWt + (size_t)l * GD * GD, GD, GD, tg);
            size_t td = (size_t)GFF * GD; // dense_W[l]: K=512, N=2048 -> Bt[2048,512]
            transpose_bf16_kernel<<<(int)((td + 255) / 256), blk, 0, stream>>>(
                dense_W + (size_t)l * GD * GFF, denseWt + (size_t)l * GFF * GD, GD, GFF, td);
        }
    }

    // head: x = node_states @ head_W + head_b
    gemm_bf16_kernel<<<dim3(GD / 64, gmM), blk, 0, stream>>>(
        node_bf, headWt, head_b, x, GN, F_IN, GD, 0);

    const size_t zcount = nD + 2 * (size_t)GN * HEADS; // attn + denom + menc (contig)
    const int lnGrid   = (GN + 7) / 8;
    const int edgeGrid = (GE + 7) / 8;

    for (int l = 0; l < G_L; ++l) {
        const float* gg = normg_g + (size_t)l * GD;
        const float* gb = normg_b + (size_t)l * GD;
        const __bf16* gW = gatWt  + (size_t)l * GD * GD;
        const float* ga = gat_a   + (size_t)l * HEADS * GU;
        const float* gbi= gat_b   + (size_t)l * GD;
        const float* dg = normd_g + (size_t)l * GD;
        const float* db = normd_b + (size_t)l * GD;
        const __bf16* dW = denseWt + (size_t)l * GFF * GD;
        const float* dbi= dense_b + (size_t)l * GFF;

        // xg = LN(x) (fp32 + bf16 copy)
        layernorm_kernel<<<lnGrid, blk, 0, stream>>>(x, xg, nullptr, xg_bf, gg, gb, GN);
        // h = xg @ gat_W + gat_b
        gemm_bf16_kernel<<<dim3(GD / 64, gmM), blk, 0, stream>>>(
            xg_bf, gW, gbi, hbuf, GN, GD, GD, 0);
        // zero attn / denom / menc
        fill_zero_kernel<<<(int)((zcount + 255) / 256), blk, 0, stream>>>(attn, zcount);
        // scores + segment max
        edge_scores_kernel<<<edgeGrid, blk, 0, stream>>>(hbuf, edges, ga, scores, menc, GE);
        // exp + segment sum
        edge_exp_kernel<<<(GE * HEADS + 255) / 256, blk, 0, stream>>>(scores, edges, menc, denom, GE);
        // weighted scatter
        edge_scatter_kernel<<<edgeGrid, blk, 0, stream>>>(hbuf, edges, scores, denom, attn, GE);
        // x = attn + xg (residual uses layernormed input)
        add_kernel<<<(int)((nD + 255) / 256), blk, 0, stream>>>(attn, xg, x, nD);
        // xr = LN(x) -> xg (+bf16), and pre-seed x = xr for fused FFN epilogue
        layernorm_kernel<<<lnGrid, blk, 0, stream>>>(x, xg, x, xg_bf, dg, db, GN);
        // x += sum_k gelu(xg @ dense_W + dense_b)[:, k*512 + d]
        gemm_bf16_kernel<<<dim3(GFF / 64, gmM), blk, 0, stream>>>(
            xg_bf, dW, dbi, x, GN, GD, GFF, 1);
    }

    // tail on gathered rows
    tail_kernel<<<N_IDX, dim3(64), 0, stream>>>(x, indices, tail_W, tail_b, out);
}
